// MLPAttention_70033736728791
// MI455X (gfx1250) — matrix operand order repaired and checked by reading, not compiled, not run on hardware
//
#include <hip/hip_runtime.h>
#include <hip/hip_bf16.h>
#include <math.h>

// Problem constants (from reference setup_inputs)
#define BB 32
#define LL 2048
#define MM 512
#define HH 512
#define AA 256
#define NEG_NUM -10000.0f

typedef __attribute__((ext_vector_type(16))) __bf16 v16bf;
typedef __attribute__((ext_vector_type(8)))  float  v8f;

union BFrag {
    v16bf v;
    uint4 q[2];
};

__device__ __forceinline__ unsigned short f2bf_rne(float f) {
    unsigned u = __float_as_uint(f);
    unsigned r = u + 0x7FFFu + ((u >> 16) & 1u);
    return (unsigned short)(r >> 16);
}

// ---------------------------------------------------------------------------
// Kernel 1: convert W1m (A x M slice of W1, row stride M+H) to bf16 scratch.
// ---------------------------------------------------------------------------
__global__ void k_convert_w1m(const float* __restrict__ W1,
                              unsigned short* __restrict__ w1mbf) {
    int idx = blockIdx.x * blockDim.x + threadIdx.x;   // 512 blocks * 256
    int a = idx >> 9;          // /512
    int k = idx & 511;
    w1mbf[a * MM + k] = f2bf_rne(W1[(size_t)a * (MM + HH) + k]);
}

// ---------------------------------------------------------------------------
// Kernel 2: hvec[b,a] = b1[a] + sum_h hidden[b,h] * W1[a, M+h]   (fp32)
// ---------------------------------------------------------------------------
__global__ void k_hvec(const float* __restrict__ hidden,
                       const float* __restrict__ W1,
                       const float* __restrict__ b1,
                       float* __restrict__ hvec) {
    int b = blockIdx.x;
    int a = threadIdx.x;
    const float* hb = hidden + (size_t)b * HH;
    const float* wr = W1 + (size_t)a * (MM + HH) + MM;
    float s = b1[a];
#pragma unroll 8
    for (int h = 0; h < HH; ++h) s = fmaf(hb[h], wr[h], s);
    hvec[b * AA + a] = s;
}

// ---------------------------------------------------------------------------
// Kernel 3: main fused GEMM + tanh + W2 reduction -> raw scores.
//   grid = (L/64, B), block = 256 threads = 8 waves (wave32).
//   Wave w computes proj tiles for lblocks 0..3 and ablocks {2w, 2w+1},
//   K=512 via 16 steps of v_wmma_f32_16x16x32_bf16.
// ---------------------------------------------------------------------------
__global__ void __launch_bounds__(256)
k_scores(const float* __restrict__ memory,
         const unsigned short* __restrict__ w1mbf,
         const float* __restrict__ hvec,
         const float* __restrict__ W2,
         const float* __restrict__ b2,
         float* __restrict__ scores) {
    __shared__ __align__(16) unsigned short smem[64 * 264]; // 64 rows, K-chunk 256, pad 8 (stride 264)
    __shared__ float sscore[64];

    const int tid  = threadIdx.x;
    const int lane = tid & 31;
    const int wv   = tid >> 5;         // 0..7
    const int hi   = lane >> 4;        // 0/1 half-wave
    const int n    = lane & 15;

    const int lt = blockIdx.x;         // l-tile (64 rows)
    const int b  = blockIdx.y;

    const float* memb = memory + ((size_t)b * LL + (size_t)lt * 64) * MM;

    if (tid < 64) sscore[tid] = 0.0f;

    v8f acc[4][2];
#pragma unroll
    for (int lb = 0; lb < 4; ++lb) {
        acc[lb][0] = (v8f)0.0f;
        acc[lb][1] = (v8f)0.0f;
    }

    const int a0 = wv * 32 + n;        // first  a column for this lane
    const int a1 = a0 + 16;            // second a column

#pragma unroll
    for (int kc = 0; kc < 2; ++kc) {
        __syncthreads();
        // ---- stage 64 x 256 fp32 -> bf16 into LDS (float4 loads, 8B stores)
        {
            const int kbase = kc * 256;
#pragma unroll
            for (int it = 0; it < 16; ++it) {
                int idx = tid + it * 256;       // float4 index, 4096 total
                int row = idx >> 6;             // 64 float4 per row
                int c4  = idx & 63;
                const float4 src = *(const float4*)(memb + (size_t)row * MM + kbase + c4 * 4);
                ushort4 d;
                d.x = f2bf_rne(src.x);
                d.y = f2bf_rne(src.y);
                d.z = f2bf_rne(src.z);
                d.w = f2bf_rne(src.w);
                *(ushort4*)&smem[row * 264 + c4 * 4] = d;
            }
        }
        __syncthreads();

#pragma unroll
        for (int ks = 0; ks < 8; ++ks) {
            const int kg = kc * 256 + ks * 32;       // global K base of this step
            // ---- B fragments (K x N = 32 x 16), contiguous k, same as A: lane n holds col a,
            //      elements j = W1m[a][kbase + j], kbase = kg + hi*16.
            BFrag fb0, fb1;
            {
                const unsigned short* p0 = w1mbf + (size_t)a0 * MM + kg + hi * 16;
                fb0.q[0] = *(const uint4*)p0;
                fb0.q[1] = *(const uint4*)(p0 + 8);
                const unsigned short* p1 = w1mbf + (size_t)a1 * MM + kg + hi * 16;
                fb1.q[0] = *(const uint4*)p1;
                fb1.q[1] = *(const uint4*)(p1 + 8);
            }
            const int kl = ks * 32;                  // K base within LDS chunk
#pragma unroll
            for (int lb = 0; lb < 4; ++lb) {
                // ---- A fragment (M x K = 16 x 32): lane m = n,
                //      elems j -> K = kl + hi*16 + j
                BFrag fa;
                const unsigned short* pa = &smem[(lb * 16 + n) * 264 + kl + hi * 16];
                fa.q[0] = *(const uint4*)pa;
                fa.q[1] = *(const uint4*)(pa + 8);

                acc[lb][0] = __builtin_amdgcn_wmma_f32_16x16x32_bf16(
                    false, fa.v, false, fb0.v, (short)0, acc[lb][0], false, false);
                acc[lb][1] = __builtin_amdgcn_wmma_f32_16x16x32_bf16(
                    false, fa.v, false, fb1.v, (short)0, acc[lb][1], false, false);
            }
        }
    }

    // ---- epilogue: + hvec, tanh, * W2, reduce over a (16 lanes), LDS atomic
    const float hv0 = hvec[b * AA + a0];
    const float hv1 = hvec[b * AA + a1];
    const float w20 = W2[a0];
    const float w21 = W2[a1];

#pragma unroll
    for (int lb = 0; lb < 4; ++lb) {
#pragma unroll
        for (int r = 0; r < 8; ++r) {
            float t0 = tanhf(acc[lb][0][r] + hv0) * w20;
            float t1 = tanhf(acc[lb][1][r] + hv1) * w21;
            float s = t0 + t1;
            // butterfly within each 16-lane half (a-dimension)
            s += __shfl_xor(s, 8);
            s += __shfl_xor(s, 4);
            s += __shfl_xor(s, 2);
            s += __shfl_xor(s, 1);
            if (n == 0) {
                int lrow = lb * 16 + hi * 8 + r;     // D layout: VGPR r -> M=hi*8+r
                atomicAdd(&sscore[lrow], s);
            }
        }
    }
    __syncthreads();

    if (tid < 64) {
        scores[(size_t)b * LL + lt * 64 + tid] = sscore[tid] + b2[0];
    }
}

// ---------------------------------------------------------------------------
// Kernel 4: masked softmax over L per batch. Writes p_memory to output.
// ---------------------------------------------------------------------------
__global__ void __launch_bounds__(256)
k_softmax(const float* __restrict__ scores,
          const int* __restrict__ mask,
          float* __restrict__ out_p) {
    __shared__ float red[256];
    const int b = blockIdx.x;
    const int tid = threadIdx.x;

    float v[8];
    float mx = -3.4e38f;
#pragma unroll
    for (int i = 0; i < 8; ++i) {
        int l = i * 256 + tid;
        float s = scores[(size_t)b * LL + l];
        s = (mask[(size_t)b * LL + l] == 0) ? NEG_NUM : s;
        v[i] = s;
        mx = fmaxf(mx, s);
    }
    red[tid] = mx;
    __syncthreads();
    for (int off = 128; off > 0; off >>= 1) {
        if (tid < off) red[tid] = fmaxf(red[tid], red[tid + off]);
        __syncthreads();
    }
    mx = red[0];
    __syncthreads();

    float sum = 0.0f;
#pragma unroll
    for (int i = 0; i < 8; ++i) {
        float e = __expf(v[i] - mx);
        v[i] = e;
        sum += e;
    }
    red[tid] = sum;
    __syncthreads();
    for (int off = 128; off > 0; off >>= 1) {
        if (tid < off) red[tid] += red[tid + off];
        __syncthreads();
    }
    const float inv = 1.0f / red[0];
#pragma unroll
    for (int i = 0; i < 8; ++i) {
        out_p[(size_t)b * LL + i * 256 + tid] = v[i] * inv;
    }
}

// ---------------------------------------------------------------------------
// Kernel 5: expected[b,m] = sum_l p[b,l] * memory[b,l,m]
//   grid = (M/256, B); p staged in LDS, coalesced 1KB rows (L2 resident).
// ---------------------------------------------------------------------------
__global__ void __launch_bounds__(256)
k_expected(const float* __restrict__ memory,
           const float* __restrict__ p,
           float* __restrict__ out_em) {
    __shared__ float sp[LL];
    const int b = blockIdx.y;
    const int tid = threadIdx.x;
#pragma unroll
    for (int i = 0; i < 8; ++i) sp[i * 256 + tid] = p[(size_t)b * LL + i * 256 + tid];
    __syncthreads();

    const int m = blockIdx.x * 256 + tid;
    const float* mb = memory + (size_t)b * LL * MM + m;
    float acc = 0.0f;
#pragma unroll 8
    for (int l = 0; l < LL; ++l) {
        acc = fmaf(sp[l], mb[(size_t)l * MM], acc);
    }
    out_em[(size_t)b * MM + m] = acc;
}

// ---------------------------------------------------------------------------
extern "C" void kernel_launch(void* const* d_in, const int* in_sizes, int n_in,
                              void* d_out, int out_size, void* d_ws, size_t ws_size,
                              hipStream_t stream) {
    const float* memory = (const float*)d_in[0];
    const float* hidden = (const float*)d_in[1];
    const int*   mask   = (const int*)d_in[2];
    const float* W1     = (const float*)d_in[3];
    const float* b1     = (const float*)d_in[4];
    const float* W2     = (const float*)d_in[5];
    const float* b2     = (const float*)d_in[6];

    float* out_em = (float*)d_out;                     // (B, M)
    float* out_p  = (float*)d_out + (size_t)BB * MM;   // (B, L)

    // workspace layout
    char* ws = (char*)d_ws;
    float*          hvec   = (float*)ws;                               // 32*256 f32   = 32 KB
    unsigned short* w1mbf  = (unsigned short*)(ws + 32768);            // 256*512 u16  = 256 KB
    float*          scores = (float*)(ws + 32768 + 262144);            // 32*2048 f32  = 256 KB

    k_convert_w1m<<<512, 256, 0, stream>>>(W1, w1mbf);
    k_hvec<<<BB, AA, 0, stream>>>(hidden, W1, b1, hvec);
    k_scores<<<dim3(LL / 64, BB), 256, 0, stream>>>(memory, w1mbf, hvec, W2, b2, scores);
    k_softmax<<<BB, 256, 0, stream>>>(scores, mask, out_p);
    k_expected<<<dim3(MM / 256, BB), 256, 0, stream>>>(memory, out_p, out_em);
}
